// RSSM_34136400068910
// MI455X (gfx1250) — compile-verified, matrix operand order checked by reading, not run on hardware
//
#include <hip/hip_runtime.h>

typedef _Float16 h16;
typedef __attribute__((ext_vector_type(16))) _Float16 v16h;
typedef __attribute__((ext_vector_type(8)))  _Float16 v8h;
typedef __attribute__((ext_vector_type(8)))  float    v8f;

#define B_   256
#define T_   50
#define E_   1536
#define A_   6
#define S_   1024
#define D_   1024
#define N_   1024
#define G3_  3072
#define KGE1 1056   // S_+A_=1030 padded to multiple of 32
#define KPO1 2560   // D_+E_
#define OUTROW ((long long)T_ * 1024)

// ---------------------------------------------------------------------------
// fp32 weight -> f16 [N, Kpad] row-major (optionally transposing [K,N] input)
// ---------------------------------------------------------------------------
__global__ void k_convert_w(const float* __restrict__ in, h16* __restrict__ out,
                            int N, int K, int Kpad, int trans) {
  int i = blockIdx.x * blockDim.x + threadIdx.x;
  if (i >= N * Kpad) return;
  int n = i / Kpad, k = i % Kpad;
  float v = 0.f;
  if (k < K) v = trans ? in[(long long)k * N + n] : in[(long long)n * K + k];
  out[i] = (h16)v;
}

// zero initial recurrent state
__global__ void k_init_state(h16* stoch16, float* deter32, h16* deter16) {
  int i = blockIdx.x * blockDim.x + threadIdx.x;
  if (i >= B_ * D_) return;
  stoch16[i] = (h16)0.f; deter32[i] = 0.f; deter16[i] = (h16)0.f;
}

// h = deter * nonterm, as f16 GEMM operand
__global__ void k_prep_h(const float* __restrict__ deter32,
                         const float* __restrict__ nonterms, int t,
                         h16* __restrict__ h16o) {
  int i = blockIdx.x * blockDim.x + threadIdx.x;
  if (i >= B_ * D_) return;
  int b = i >> 10;
  h16o[i] = (h16)(deter32[i] * nonterms[b * T_ + t]);
}

// ge_in = [stoch*nt, action*nt, pad] as f16 [B, 1056]
__global__ void k_build_gein(const h16* __restrict__ stoch16,
                             const float* __restrict__ action,
                             const float* __restrict__ nonterms, int t,
                             h16* __restrict__ gein) {
  int i = blockIdx.x * blockDim.x + threadIdx.x;
  if (i >= B_ * KGE1) return;
  int b = i / KGE1, j = i % KGE1;
  float nt = nonterms[b * T_ + t];
  float v = 0.f;
  if (j < S_)              v = (float)stoch16[b * S_ + j] * nt;
  else if (j < S_ + A_)    v = action[((long long)b * T_ + t) * A_ + (j - S_)] * nt;
  gein[i] = (h16)v;
}

// po_in = [new_deter, obs_embed] as f16 [B, 2560]
__global__ void k_build_poin(const h16* __restrict__ deter16,
                             const float* __restrict__ obs, int t,
                             h16* __restrict__ poin) {
  int i = blockIdx.x * blockDim.x + threadIdx.x;
  if (i >= B_ * KPO1) return;
  int b = i / KPO1, j = i % KPO1;
  h16 v;
  if (j < D_) v = deter16[b * D_ + j];
  else        v = (h16)obs[((long long)b * T_ + t) * E_ + (j - D_)];
  poin[i] = v;
}

// ---------------------------------------------------------------------------
// WMMA GEMM: C[M=256, N] = act( A[M,K]f16 @ Bt[N,K]f16^T + bias )
// Block = 8 waves -> 32(M) x 256(N) tile. The FULL 32-row A panel (32*K halves,
// <= 160 KB) is staged into LDS once via CDNA5 async copy
// (global_load_async_to_lds_b128, ASYNCcnt), one wait + one barrier. The K loop
// is then barrier-free: per 32-K chunk each wave issues 2 ds_load_b128 (A frag,
// reused 4x) + 8 global_load_b128 (weights, L2-hot) + 4 independent WMMAs.
// ---------------------------------------------------------------------------
__global__ __launch_bounds__(256)
void gemm_wmma_f16(const h16* __restrict__ Amat, int lda,
                   const h16* __restrict__ Bt, int ldb,
                   const float* __restrict__ bias,
                   float* __restrict__ C32, long long ldc32,
                   h16* __restrict__ C16, int ldc16,
                   int K, int act) {
  extern __shared__ h16 As[];          // 32 * K halves, row-major (== global layout)

  const int tid  = threadIdx.x;
  const int wave = tid >> 5, lane = tid & 31;
  const int bm   = blockIdx.y * 32;
  const int bn   = blockIdx.x * 256;
  const int wm   = (wave >> 2) * 16;   // 0 | 16
  const int wn   = (wave & 3) * 64;    // 0,64,128,192 : 4 N-tiles per wave
  const int l15  = lane & 15;
  const int hi   = lane >> 4;          // K-half select per ISA fragment layout

  // ---- async-stage entire A panel: 32 contiguous rows (lda == K) ----------
  {
    const char* panel = (const char*)(Amat + (long long)bm * lda);
    const int chunks = 4 * K;          // (32 rows * K halves * 2B) / 16B
    for (int c = tid; c < chunks; c += 256) {
      unsigned    la = (unsigned)(c * 16);          // LDS byte address
      const char* ga = panel + (size_t)c * 16;      // global byte address
      asm volatile("global_load_async_to_lds_b128 %0, %1, off"
                   :: "v"(la), "v"(ga) : "memory");
    }
    asm volatile("s_wait_asynccnt 0x0" ::: "memory");
    __syncthreads();
  }

  const h16* arow = As + (long long)(wm + l15) * K + hi * 8;
  const h16* bc0  = Bt + (long long)(bn + wn +  0 + l15) * ldb + hi * 8;
  const h16* bc1  = Bt + (long long)(bn + wn + 16 + l15) * ldb + hi * 8;
  const h16* bc2  = Bt + (long long)(bn + wn + 32 + l15) * ldb + hi * 8;
  const h16* bc3  = Bt + (long long)(bn + wn + 48 + l15) * ldb + hi * 8;

  v8f acc0 = {}, acc1 = {}, acc2 = {}, acc3 = {};

  for (int k0 = 0; k0 < K; k0 += 32) {
    // A fragment (16x32 f16) from LDS: lane<16 -> K 0..7 & 16..23; lane>=16 -> +8
    v8h alo = *(const v8h*)(arow + k0);
    v8h ahi = *(const v8h*)(arow + k0 + 16);
    v16h a  = __builtin_shufflevector(alo, ahi, 0,1,2,3,4,5,6,7,8,9,10,11,12,13,14,15);

    v8h b0lo = *(const v8h*)(bc0 + k0), b0hi = *(const v8h*)(bc0 + k0 + 16);
    v8h b1lo = *(const v8h*)(bc1 + k0), b1hi = *(const v8h*)(bc1 + k0 + 16);
    v8h b2lo = *(const v8h*)(bc2 + k0), b2hi = *(const v8h*)(bc2 + k0 + 16);
    v8h b3lo = *(const v8h*)(bc3 + k0), b3hi = *(const v8h*)(bc3 + k0 + 16);
    __builtin_prefetch(bc0 + k0 + 32, 0, 0);
    __builtin_prefetch(bc2 + k0 + 32, 0, 0);

    v16h b0 = __builtin_shufflevector(b0lo, b0hi, 0,1,2,3,4,5,6,7,8,9,10,11,12,13,14,15);
    v16h b1 = __builtin_shufflevector(b1lo, b1hi, 0,1,2,3,4,5,6,7,8,9,10,11,12,13,14,15);
    v16h b2 = __builtin_shufflevector(b2lo, b2hi, 0,1,2,3,4,5,6,7,8,9,10,11,12,13,14,15);
    v16h b3 = __builtin_shufflevector(b3lo, b3hi, 0,1,2,3,4,5,6,7,8,9,10,11,12,13,14,15);

    // 4 independent WMMAs -> back-to-back issue on the matrix pipe
    acc0 = __builtin_amdgcn_wmma_f32_16x16x32_f16(false, a, false, b0, (short)0, acc0, false, false);
    acc1 = __builtin_amdgcn_wmma_f32_16x16x32_f16(false, a, false, b1, (short)0, acc1, false, false);
    acc2 = __builtin_amdgcn_wmma_f32_16x16x32_f16(false, a, false, b2, (short)0, acc2, false, false);
    acc3 = __builtin_amdgcn_wmma_f32_16x16x32_f16(false, a, false, b3, (short)0, acc3, false, false);
  }

  v8f accs[4] = {acc0, acc1, acc2, acc3};
#pragma unroll
  for (int nt = 0; nt < 4; ++nt) {
    const int col = bn + wn + nt * 16 + l15;
    const float bv = bias ? bias[col] : 0.f;
#pragma unroll
    for (int r = 0; r < 8; ++r) {
      int row = bm + wm + r + hi * 8;    // C/D layout: VGPR r -> M=r / M=r+8
      float v = accs[nt][r] + bv;
      if (act == 1) v = v > 0.f ? v : (__expf(v) - 1.f);   // ELU
      if (C32) C32[(long long)row * ldc32 + col] = v;
      if (C16) C16[(long long)row * ldc16 + col] = (h16)v;
    }
  }
}

// ---------------------------------------------------------------------------
// fused GRU gates: new_h = (1-z)*n + z*h ; writes state + output slots 2 & 5
// ---------------------------------------------------------------------------
__global__ void k_gru(const float* __restrict__ gi, const float* __restrict__ gh,
                      float* __restrict__ deter32, h16* __restrict__ deter16,
                      const float* __restrict__ nonterms, int t,
                      float* __restrict__ out) {
  int i = blockIdx.x * blockDim.x + threadIdx.x;
  if (i >= B_ * D_) return;
  int b = i >> 10, j = i & 1023;
  const float* gib = gi + (long long)b * G3_;
  const float* ghb = gh + (long long)b * G3_;
  float r = 1.f / (1.f + __expf(-(gib[j]          + ghb[j])));
  float z = 1.f / (1.f + __expf(-(gib[j + 1024]   + ghb[j + 1024])));
  float n = tanhf(gib[j + 2048] + r * ghb[j + 2048]);
  float h = deter32[i] * nonterms[b * T_ + t];
  float hn = (1.f - z) * n + z * h;
  deter32[i] = hn;
  deter16[i] = (h16)hn;
  out[((long long)(2 * B_ + b) * T_ + t) * 1024 + j] = hn;
  out[((long long)(5 * B_ + b) * T_ + t) * 1024 + j] = hn;
}

// ---------------------------------------------------------------------------
// straight-through categorical sample: one-hot(argmax(logit + gumbel))
// ---------------------------------------------------------------------------
__device__ inline unsigned wang(unsigned s) {
  s = (s ^ 61u) ^ (s >> 16); s *= 9u; s ^= s >> 4; s *= 0x27d4eb2du; s ^= s >> 15;
  return s;
}

__global__ void k_sample(float* __restrict__ out, int t, int f,
                         h16* __restrict__ stoch16) {
  int idx = blockIdx.x * blockDim.x + threadIdx.x;   // B_*CAT = 8192
  if (idx >= B_ * 32) return;
  int b = idx >> 5, cat = idx & 31;
  long long lbase = ((long long)(f * B_ + b) * T_ + t) * 1024 + cat * 32;
  int best = 0; float bestv = -3.4e38f;
#pragma unroll
  for (int c = 0; c < 32; ++c) {
    unsigned hsh = wang(wang(idx * 1021u + c * 131u + 0x9E3779B9u) ^
                        (t * 2654435761u + f * 40503u + 7u));
    float u = (float)(hsh >> 8) * (1.f / 16777216.f) + 1e-7f;
    float g = -__logf(-__logf(u));
    float v = out[lbase + c] + g;
    if (v > bestv) { bestv = v; best = c; }
  }
  long long obase = ((long long)((f + 1) * B_ + b) * T_ + t) * 1024 + cat * 32;
#pragma unroll
  for (int c = 0; c < 32; ++c) {
    float oh = (c == best) ? 1.f : 0.f;
    out[obase + c] = oh;
    if (stoch16) stoch16[b * S_ + cat * 32 + c] = (h16)oh;
  }
}

// ---------------------------------------------------------------------------
extern "C" void kernel_launch(void* const* d_in, const int* in_sizes, int n_in,
                              void* d_out, int out_size, void* d_ws, size_t ws_size,
                              hipStream_t stream) {
  const float* obs   = (const float*)d_in[0];
  const float* act   = (const float*)d_in[1];
  const float* nonT  = (const float*)d_in[2];
  const float* ge_w1 = (const float*)d_in[3];
  const float* ge_b1 = (const float*)d_in[4];
  const float* ge_w2 = (const float*)d_in[5];
  const float* ge_b2 = (const float*)d_in[6];
  const float* wih   = (const float*)d_in[7];
  const float* whh   = (const float*)d_in[8];
  const float* bih   = (const float*)d_in[9];
  const float* bhh   = (const float*)d_in[10];
  const float* pr_w1 = (const float*)d_in[11];
  const float* pr_b1 = (const float*)d_in[12];
  const float* pr_w2 = (const float*)d_in[13];
  const float* pr_b2 = (const float*)d_in[14];
  const float* po_w1 = (const float*)d_in[15];
  const float* po_b1 = (const float*)d_in[16];
  const float* po_w2 = (const float*)d_in[17];
  const float* po_b2 = (const float*)d_in[18];
  float* out = (float*)d_out;

  char*  ws  = (char*)d_ws;
  size_t off = 0;
  auto take = [&](size_t bytes) -> char* {
    char* p = ws + off; off = (off + bytes + 255) & ~(size_t)255; return p;
  };

  // f16 weights, all pre-transposed to [N, K] row-major
  h16* Wge1 = (h16*)take((size_t)N_  * KGE1 * 2);
  h16* Wge2 = (h16*)take((size_t)D_  * N_   * 2);
  h16* Wih  = (h16*)take((size_t)G3_ * D_   * 2);
  h16* Whh  = (h16*)take((size_t)G3_ * D_   * 2);
  h16* Wpr1 = (h16*)take((size_t)N_  * D_   * 2);
  h16* Wpr2 = (h16*)take((size_t)S_  * N_   * 2);
  h16* Wpo1 = (h16*)take((size_t)N_  * KPO1 * 2);
  h16* Wpo2 = (h16*)take((size_t)S_  * N_   * 2);
  // activations / state
  h16*   gein    = (h16*)take((size_t)B_ * KGE1 * 2);
  h16*   h1      = (h16*)take((size_t)B_ * N_ * 2);
  h16*   xe      = (h16*)take((size_t)B_ * D_ * 2);   // gru_embed
  h16*   hstate  = (h16*)take((size_t)B_ * D_ * 2);   // deter*nt (f16)
  float* deter32 = (float*)take((size_t)B_ * D_ * 4);
  h16*   deter16 = (h16*)take((size_t)B_ * D_ * 2);
  float* gibuf   = (float*)take((size_t)B_ * G3_ * 4);
  float* ghbuf   = (float*)take((size_t)B_ * G3_ * 4);
  h16*   ph1     = (h16*)take((size_t)B_ * N_ * 2);
  h16*   qh1     = (h16*)take((size_t)B_ * N_ * 2);
  h16*   poin    = (h16*)take((size_t)B_ * KPO1 * 2);
  h16*   stoch16 = (h16*)take((size_t)B_ * S_ * 2);
  (void)ws_size; (void)in_sizes; (void)n_in; (void)out_size;

  auto cvt = [&](const float* src, h16* dst, int N, int K, int Kpad, int tr) {
    int n = N * Kpad;
    k_convert_w<<<(n + 255) / 256, 256, 0, stream>>>(src, dst, N, K, Kpad, tr);
  };
  cvt(ge_w1, Wge1, N_,  S_ + A_, KGE1, 1);
  cvt(ge_w2, Wge2, D_,  N_,      N_,   1);
  cvt(wih,   Wih,  G3_, D_,      D_,   0);   // already [3D, D] = [N, K]
  cvt(whh,   Whh,  G3_, D_,      D_,   0);
  cvt(pr_w1, Wpr1, N_,  D_,      D_,   1);
  cvt(pr_w2, Wpr2, S_,  N_,      N_,   1);
  cvt(po_w1, Wpo1, N_,  KPO1,    KPO1, 1);
  cvt(po_w2, Wpo2, S_,  N_,      N_,   1);

  k_init_state<<<(B_ * D_ + 255) / 256, 256, 0, stream>>>(stoch16, deter32, deter16);

  auto gemm = [&](const h16* A, int lda, const h16* Bt, int ldb, const float* bias,
                  float* C32, long long ldc32, h16* C16, int ldc16,
                  int N, int K, int actf) {
    dim3 grid(N / 256, B_ / 32);
    size_t shmem = (size_t)32 * K * 2;   // full A panel in LDS (<=160 KB)
    gemm_wmma_f16<<<grid, 256, shmem, stream>>>(A, lda, Bt, ldb, bias,
                                                C32, ldc32, C16, ldc16, K, actf);
  };

  const int n1   = (B_ * D_   + 255) / 256;
  const int nge  = (B_ * KGE1 + 255) / 256;
  const int npo  = (B_ * KPO1 + 255) / 256;

  for (int t = 0; t < T_; ++t) {
    // inputs for this step
    k_prep_h   <<<n1,  256, 0, stream>>>(deter32, nonT, t, hstate);
    k_build_gein<<<nge, 256, 0, stream>>>(stoch16, act, nonT, t, gein);

    // gru_embed MLP
    gemm(gein, KGE1, Wge1, KGE1, ge_b1, nullptr, 0, h1, N_, N_, KGE1, 1);
    gemm(h1,   N_,   Wge2, N_,   ge_b2, nullptr, 0, xe, D_, D_, N_,   0);

    // GRU gates
    gemm(xe,     D_, Wih, D_, bih, gibuf, G3_, nullptr, 0, G3_, D_, 0);
    gemm(hstate, D_, Whh, D_, bhh, ghbuf, G3_, nullptr, 0, G3_, D_, 0);
    k_gru<<<n1, 256, 0, stream>>>(gibuf, ghbuf, deter32, deter16, nonT, t, out);

    // prior MLP -> logits straight into d_out slot 0
    gemm(deter16, D_, Wpr1, D_, pr_b1, nullptr, 0, ph1, N_, N_, D_, 1);
    gemm(ph1, N_, Wpr2, N_, pr_b2,
         out + (long long)t * 1024, OUTROW, nullptr, 0, S_, N_, 0);
    k_sample<<<(B_ * 32 + 255) / 256, 256, 0, stream>>>(out, t, 0, nullptr);

    // posterior MLP -> logits into d_out slot 3
    k_build_poin<<<npo, 256, 0, stream>>>(deter16, obs, t, poin);
    gemm(poin, KPO1, Wpo1, KPO1, po_b1, nullptr, 0, qh1, N_, N_, KPO1, 1);
    gemm(qh1, N_, Wpo2, N_, po_b2,
         out + ((long long)3 * B_ * T_ + t) * 1024, OUTROW, nullptr, 0, S_, N_, 0);
    k_sample<<<(B_ * 32 + 255) / 256, 256, 0, stream>>>(out, t, 3, stoch16);
  }
}
